// TutelAdapter_47579647705116
// MI455X (gfx1250) — matrix operand order repaired and checked by reading, not compiled, not run on hardware
//
#include <hip/hip_runtime.h>
#include <hip/hip_bf16.h>

// ---------------------------------------------------------------------------
// MoE (tutel-style) forward for MI455X / gfx1250.
// Heavy math (2x grouped GEMMs, 6.9e11 FLOPs) runs on v_wmma_f32_16x16x32_f16
// with f32 accumulation. f32 inputs/weights are cast to f16 once per launch
// (fc2 is cast+transposed via LDS so both GEMMs are K-contiguous for WMMA
// fragment loads). Wave tile 64xN (MI=4) to raise FLOPs per L2 byte.
// ---------------------------------------------------------------------------

#define NTOK   16384      // B*S
#define DDIM   1024
#define NEXP   8
#define HDIM   4096
#define TOPK   2
#define CAP    5120       // 2 * int(1.25 * 2048)
#define ECAP   (NEXP * CAP)   // 40960
#define EPSV   1.1920929e-07f

typedef _Float16 v16h __attribute__((ext_vector_type(16)));
typedef _Float16 v8h  __attribute__((ext_vector_type(8)));
typedef _Float16 v4h  __attribute__((ext_vector_type(4)));
typedef float    v8f  __attribute__((ext_vector_type(8)));

// ---------------------------------------------------------------------------
// 1) Gate: scores = softmax(x @ wg), top-2, renormalize. One wave per token.
// ---------------------------------------------------------------------------
__global__ __launch_bounds__(256) void gate_topk_kernel(
    const float* __restrict__ x, const float* __restrict__ wg,
    int* __restrict__ idx, float* __restrict__ gates) {
  const int wave = threadIdx.x >> 5;
  const int lane = threadIdx.x & 31;
  const int n = blockIdx.x * 8 + wave;
  const float* xr = x + (size_t)n * DDIM;

  float acc[NEXP];
#pragma unroll
  for (int e = 0; e < NEXP; ++e) acc[e] = 0.f;

  for (int d = lane; d < DDIM; d += 32) {
    const float xv = xr[d];
    const float* w = wg + (size_t)d * NEXP;
#pragma unroll
    for (int e = 0; e < NEXP; ++e) acc[e] += xv * w[e];
  }
#pragma unroll
  for (int e = 0; e < NEXP; ++e) {
#pragma unroll
    for (int off = 16; off >= 1; off >>= 1) acc[e] += __shfl_xor(acc[e], off, 32);
  }
  if (lane == 0) {
    float m = acc[0];
#pragma unroll
    for (int e = 1; e < NEXP; ++e) m = fmaxf(m, acc[e]);
    float s = 0.f, p[NEXP];
#pragma unroll
    for (int e = 0; e < NEXP; ++e) { p[e] = __expf(acc[e] - m); s += p[e]; }
    const float inv = 1.f / s;
#pragma unroll
    for (int e = 0; e < NEXP; ++e) p[e] *= inv;
    int i0 = 0;
#pragma unroll
    for (int e = 1; e < NEXP; ++e) if (p[e] > p[i0]) i0 = e;
    int i1 = (i0 == 0) ? 1 : 0;
#pragma unroll
    for (int e = 0; e < NEXP; ++e) if (e != i0 && p[e] > p[i1]) i1 = e;
    const float g0 = p[i0], g1 = p[i1];
    const float denom = fmaxf(g0 + g1, EPSV);
    idx[n * 2 + 0] = i0;  idx[n * 2 + 1] = i1;
    gates[n * 2 + 0] = g0 / denom;  gates[n * 2 + 1] = g1 / denom;
  }
}

// ---------------------------------------------------------------------------
// 2) Deterministic per-(k,expert) running slot index via single-wave ballot
//    scan (tutel cumsum semantics, token order preserved).
// ---------------------------------------------------------------------------
__global__ void scan_kernel(const int* __restrict__ idx, int* __restrict__ loc,
                            int* __restrict__ counts) {
  const int lane = threadIdx.x;  // 32 threads, 1 wave
  int base0[NEXP], base1[NEXP];
#pragma unroll
  for (int e = 0; e < NEXP; ++e) { base0[e] = 0; base1[e] = 0; }
  const unsigned long long lmask = (1ull << lane) - 1ull;

  for (int t0 = 0; t0 < NTOK; t0 += 32) {
    const int n = t0 + lane;
    const int e0 = idx[n * 2 + 0];
    const int e1 = idx[n * 2 + 1];
#pragma unroll
    for (int e = 0; e < NEXP; ++e) {
      unsigned long long b = __ballot(e0 == e);
      if (e0 == e) loc[n * 2 + 0] = base0[e] + __popcll(b & lmask);
      base0[e] += __popcll(b);
    }
#pragma unroll
    for (int e = 0; e < NEXP; ++e) {
      unsigned long long b = __ballot(e1 == e);
      if (e1 == e) loc[n * 2 + 1] = base1[e] + __popcll(b & lmask);
      base1[e] += __popcll(b);
    }
  }
  if (lane == 0) {
#pragma unroll
    for (int e = 0; e < NEXP; ++e) { counts[e] = base0[e]; counts[NEXP + e] = base1[e]; }
  }
}

// ---------------------------------------------------------------------------
// 3) Apply k-major offsets + capacity drop -> slot position.
// ---------------------------------------------------------------------------
__global__ __launch_bounds__(256) void pos_kernel(
    const int* __restrict__ idx, const int* __restrict__ loc,
    const int* __restrict__ counts, float* __restrict__ gates,
    int* __restrict__ pos) {
  const int i = blockIdx.x * blockDim.x + threadIdx.x;  // i = n*2 + k
  if (i >= NTOK * TOPK) return;
  const int k = i & 1;
  const int e = idx[i];
  const int off = (k == 0) ? 0 : counts[e];  // exclusive cumsum over k
  const int l = loc[i] + off;
  if (l < CAP) {
    pos[i] = e * CAP + l;
  } else {
    pos[i] = ECAP;      // dummy (dropped)
    gates[i] = 0.f;
  }
}

// ---------------------------------------------------------------------------
// Utility fills / casts
// ---------------------------------------------------------------------------
__global__ __launch_bounds__(256) void fill_zero_u4(uint4* __restrict__ p, size_t n4) {
  const size_t stride = (size_t)gridDim.x * blockDim.x;
  for (size_t i = blockIdx.x * (size_t)blockDim.x + threadIdx.x; i < n4; i += stride)
    p[i] = make_uint4(0u, 0u, 0u, 0u);
}

__global__ __launch_bounds__(256) void f32_to_f16_kernel(
    const float* __restrict__ in, _Float16* __restrict__ out, size_t n) {
  const size_t stride = (size_t)gridDim.x * blockDim.x * 4;
  for (size_t j = ((size_t)blockIdx.x * blockDim.x + threadIdx.x) * 4; j < n; j += stride) {
    const float4 v = *(const float4*)(in + j);
    v4h h; h[0] = (_Float16)v.x; h[1] = (_Float16)v.y; h[2] = (_Float16)v.z; h[3] = (_Float16)v.w;
    *(v4h*)(out + j) = h;
  }
}

// fc2_w: [E,H,D] f32 -> [E,D,H] f16 (K=H made contiguous for WMMA B-frags).
// LDS-tiled 32x32 transpose, coalesced both sides.
__global__ __launch_bounds__(256) void cast_transpose_kernel(
    const float* __restrict__ in, _Float16* __restrict__ out) {
  __shared__ _Float16 tile[32][33];
  const int e = blockIdx.z;
  const int h0 = blockIdx.y * 32;
  const int d0 = blockIdx.x * 32;
  const int tx = threadIdx.x & 31;
  const int ty = threadIdx.x >> 5;  // 8 rows per pass
#pragma unroll
  for (int r = ty; r < 32; r += 8)
    tile[r][tx] = (_Float16)in[((size_t)e * HDIM + h0 + r) * DDIM + d0 + tx];
  __syncthreads();
#pragma unroll
  for (int r = ty; r < 32; r += 8)
    out[((size_t)e * DDIM + d0 + r) * HDIM + h0 + tx] = tile[tx][r];
}

// ---------------------------------------------------------------------------
// 4) Scatter kept tokens (f32 -> f16) into dispatch buffer [E*C, D].
// ---------------------------------------------------------------------------
__global__ __launch_bounds__(128) void scatter_kernel(
    const float* __restrict__ x, const int* __restrict__ pos,
    _Float16* __restrict__ disp) {
  const int i = blockIdx.x;          // n*2 + k
  const int p = pos[i];
  if (p >= ECAP) return;
  const int n = i >> 1;
  const float4* src = (const float4*)(x + (size_t)n * DDIM);
  _Float16* dst = disp + (size_t)p * DDIM;
#pragma unroll
  for (int j = threadIdx.x; j < DDIM / 4; j += 128) {
    const float4 v = src[j];
    v4h h; h[0] = (_Float16)v.x; h[1] = (_Float16)v.y; h[2] = (_Float16)v.z; h[3] = (_Float16)v.w;
    *(v4h*)(dst + 4 * j) = h;
  }
}

// ---------------------------------------------------------------------------
// 5) WMMA f16 GEMM: Out[M,N] = A[M,K] * B[N,K]^T (+bias, optional SiLU->f16).
//    Both operands K-contiguous. 8 waves/block laid out 4(M) x 2(N);
//    wave tile = (MI*16) x (NI*16); block tile = (MI*64) x (NI*32).
//    Fragment addressing: contiguous k order k = 16*(lane>>4) + s, same for A and B (wave32).
// ---------------------------------------------------------------------------
template <int MI, int NI, bool SILU>
__global__ __launch_bounds__(256) void moe_gemm_f16(
    const _Float16* __restrict__ A, const _Float16* __restrict__ B,
    const float* __restrict__ bias, void* __restrict__ Out,
    int M, int N, int K) {
  const int lane  = threadIdx.x & 31;
  const int wid   = threadIdx.x >> 5;
  const int waveM = wid & 3;
  const int waveN = wid >> 2;
  const int rowBase = blockIdx.y * (MI * 64) + waveM * (MI * 16);
  const int colBase = blockIdx.x * (NI * 32) + waveN * (NI * 16);

  // A 16x32 f16 fragment: lane holds row (lane&15); 16 contiguous K at +(lane>>4)*16.
  const int aRow = lane & 15;
  const int aK   = (lane >> 4) * 16;
  // B 32x16 fragment: lane holds col (lane&15); 16 contiguous K at +(lane>>4)*16.
  const int bCol = lane & 15;
  const int bK   = (lane >> 4) * 16;

  const _Float16* Abase = A + (size_t)rowBase * K;
  const _Float16* Bbase = B + (size_t)colBase * K;

  v8f acc[MI][NI];
#pragma unroll
  for (int mi = 0; mi < MI; ++mi)
#pragma unroll
    for (int ni = 0; ni < NI; ++ni) acc[mi][ni] = (v8f)(0.f);

  for (int k0 = 0; k0 < K; k0 += 32) {
    v16h a[MI], b[NI];
#pragma unroll
    for (int mi = 0; mi < MI; ++mi) {
      const _Float16* p = Abase + (size_t)(mi * 16 + aRow) * K + k0 + aK;
      const v8h lo = *(const v8h*)p;
      const v8h hi = *(const v8h*)(p + 8);
#pragma unroll
      for (int t = 0; t < 8; ++t) { a[mi][t] = lo[t]; a[mi][t + 8] = hi[t]; }
    }
#pragma unroll
    for (int ni = 0; ni < NI; ++ni) {
      const _Float16* p = Bbase + (size_t)(ni * 16 + bCol) * K + k0 + bK;
      b[ni] = *(const v16h*)p;
      __builtin_prefetch(p + 64, 0, 3);  // global_prefetch_b8, next K-chunks
    }
#pragma unroll
    for (int mi = 0; mi < MI; ++mi)
#pragma unroll
      for (int ni = 0; ni < NI; ++ni)
        acc[mi][ni] = __builtin_amdgcn_wmma_f32_16x16x32_f16(
            false, a[mi], false, b[ni], (short)0, acc[mi][ni], false, false);
  }

  // Epilogue. C/D frag: lane holds col (lane&15); VGPR r -> row r + (lane>>4)*8.
  const int cN = lane & 15;
  const int cM = (lane >> 4) * 8;
#pragma unroll
  for (int ni = 0; ni < NI; ++ni) {
    const int col = colBase + ni * 16 + cN;
    const float bs = bias[col];
#pragma unroll
    for (int mi = 0; mi < MI; ++mi) {
      const int row0 = rowBase + mi * 16 + cM;
#pragma unroll
      for (int r = 0; r < 8; ++r) {
        float v = acc[mi][ni][r] + bs;
        if (SILU) v = v / (1.f + __expf(-v));
        ((_Float16*)Out)[(size_t)(row0 + r) * N + col] = (_Float16)v;
      }
    }
  }
}

// ---------------------------------------------------------------------------
// 6) Combine (gather 2 slots * gates) + residual + LayerNorm. Block per token.
// ---------------------------------------------------------------------------
__global__ __launch_bounds__(256) void combine_ln_kernel(
    const float* __restrict__ x, const _Float16* __restrict__ y,
    const float* __restrict__ gates, const int* __restrict__ pos,
    const float* __restrict__ gamma, const float* __restrict__ beta,
    float* __restrict__ out) {
  const int n = blockIdx.x;
  const int p0 = pos[n * 2 + 0], p1 = pos[n * 2 + 1];
  const float g0 = gates[n * 2 + 0], g1 = gates[n * 2 + 1];
  const bool ok0 = p0 < ECAP, ok1 = p1 < ECAP;
  const _Float16* y0 = y + (size_t)p0 * DDIM;
  const _Float16* y1 = y + (size_t)p1 * DDIM;
  const float* xr = x + (size_t)n * DDIM;

  float vals[4];
  float s = 0.f, ss = 0.f;
#pragma unroll
  for (int j = 0; j < 4; ++j) {
    const int d = threadIdx.x + j * 256;
    float v = xr[d];
    if (ok0) v += g0 * (float)y0[d];
    if (ok1) v += g1 * (float)y1[d];
    vals[j] = v;
    s += v;
    ss += v * v;
  }
#pragma unroll
  for (int off = 16; off >= 1; off >>= 1) {
    s  += __shfl_xor(s, off, 32);
    ss += __shfl_xor(ss, off, 32);
  }
  __shared__ float red[18];
  const int wid = threadIdx.x >> 5, lane = threadIdx.x & 31;
  if (lane == 0) { red[wid] = s; red[8 + wid] = ss; }
  __syncthreads();
  if (threadIdx.x == 0) {
    float S = 0.f, SS = 0.f;
#pragma unroll
    for (int w = 0; w < 8; ++w) { S += red[w]; SS += red[8 + w]; }
    red[16] = S; red[17] = SS;
  }
  __syncthreads();
  const float mu = red[16] * (1.f / DDIM);
  const float var = red[17] * (1.f / DDIM) - mu * mu;
  const float rs = rsqrtf(var + 1e-5f);
#pragma unroll
  for (int j = 0; j < 4; ++j) {
    const int d = threadIdx.x + j * 256;
    out[(size_t)n * DDIM + d] = (vals[j] - mu) * rs * gamma[d] + beta[d];
  }
}

// ---------------------------------------------------------------------------
// Host launcher
// ---------------------------------------------------------------------------
extern "C" void kernel_launch(void* const* d_in, const int* in_sizes, int n_in,
                              void* d_out, int out_size, void* d_ws, size_t ws_size,
                              hipStream_t stream) {
  const float* x      = (const float*)d_in[0];  // [N, D]
  const float* wg     = (const float*)d_in[1];  // [D, E]
  const float* fc1_w  = (const float*)d_in[2];  // [E, H, D]
  const float* fc1_b  = (const float*)d_in[3];  // [E, H]
  const float* fc2_w  = (const float*)d_in[4];  // [E, H, D]
  const float* fc2_b  = (const float*)d_in[5];  // [E, D]
  const float* gamma  = (const float*)d_in[6];  // [D]
  const float* beta   = (const float*)d_in[7];  // [D]
  float* out = (float*)d_out;

  // Workspace carve (256B aligned)
  char* w = (char*)d_ws;
  auto carve = [&](size_t bytes) {
    void* p = (void*)w;
    w += (bytes + 255) & ~(size_t)255;
    return p;
  };
  int*      idx    = (int*)carve(sizeof(int) * NTOK * TOPK);
  float*    gates  = (float*)carve(sizeof(float) * NTOK * TOPK);
  int*      loc    = (int*)carve(sizeof(int) * NTOK * TOPK);
  int*      counts = (int*)carve(sizeof(int) * 16);
  int*      pos    = (int*)carve(sizeof(int) * NTOK * TOPK);
  _Float16* disp   = (_Float16*)carve(sizeof(_Float16) * (size_t)ECAP * DDIM);
  _Float16* fc1h   = (_Float16*)carve(sizeof(_Float16) * (size_t)NEXP * HDIM * DDIM);
  _Float16* fc2h   = (_Float16*)carve(sizeof(_Float16) * (size_t)NEXP * DDIM * HDIM);
  _Float16* hbuf   = (_Float16*)carve(sizeof(_Float16) * (size_t)CAP * HDIM);
  _Float16* ybuf   = (_Float16*)carve(sizeof(_Float16) * (size_t)ECAP * DDIM);
  (void)ws_size; (void)in_sizes; (void)n_in; (void)out_size;

  // 1) gate
  gate_topk_kernel<<<NTOK / 8, 256, 0, stream>>>(x, wg, idx, gates);
  // 2) deterministic scan
  scan_kernel<<<1, 32, 0, stream>>>(idx, loc, counts);
  // 3) positions + capacity drop
  pos_kernel<<<(NTOK * TOPK + 255) / 256, 256, 0, stream>>>(idx, loc, counts, gates, pos);
  // 4) zero dispatch buffer; weight casts
  fill_zero_u4<<<2048, 256, 0, stream>>>((uint4*)disp,
                                         (size_t)ECAP * DDIM * sizeof(_Float16) / 16);
  f32_to_f16_kernel<<<4096, 256, 0, stream>>>(fc1_w, fc1h, (size_t)NEXP * HDIM * DDIM);
  cast_transpose_kernel<<<dim3(DDIM / 32, HDIM / 32, NEXP), 256, 0, stream>>>(fc2_w, fc2h);
  // 5) scatter tokens
  scatter_kernel<<<NTOK * TOPK, 128, 0, stream>>>(x, pos, disp);
  // 6) per-expert grouped FFN on WMMA
  for (int e = 0; e < NEXP; ++e) {
    // GEMM1: [C,D] x [H,D]^T -> SiLU -> h [C,H] (f16); block tile 256x256
    moe_gemm_f16<4, 8, true><<<dim3(HDIM / 256, CAP / 256), 256, 0, stream>>>(
        disp + (size_t)e * CAP * DDIM, fc1h + (size_t)e * HDIM * DDIM,
        fc1_b + (size_t)e * HDIM, (void*)hbuf, CAP, HDIM, DDIM);
    // GEMM2: [C,H] x [D,H]^T -> y [C,D] (f16); block tile 256x128
    moe_gemm_f16<4, 4, false><<<dim3(DDIM / 128, CAP / 256), 256, 0, stream>>>(
        hbuf, fc2h + (size_t)e * DDIM * HDIM, fc2_b + (size_t)e * DDIM,
        (void*)(ybuf + (size_t)e * CAP * DDIM), CAP, DDIM, HDIM);
  }
  // 7) combine + residual + LayerNorm
  combine_ln_kernel<<<NTOK, 256, 0, stream>>>(x, ybuf, gates, pos, gamma, beta, out);
}
